// TransformerBlock_47665547051172
// MI455X (gfx1250) — matrix operand order repaired and checked by reading, not compiled, not run on hardware
//
#include <hip/hip_runtime.h>
#include <hip/hip_bf16.h>

typedef _Float16 v16h __attribute__((ext_vector_type(16)));
typedef _Float16 v8h  __attribute__((ext_vector_type(8)));
typedef float    v8f  __attribute__((ext_vector_type(8)));

#define D_MODEL 1024
#define DFF     4096
#define SEQ     2048
#define BATCH   2
#define NHEAD   16
#define DEPTH   64
#define NTOK    (BATCH*SEQ)

// ---------------------------------------------------------------------------
// WMMA fragment helpers (CDNA5 16x16x32 f16, wave32)
// A layout: lane 0-15 -> M=lane, elems 0..7 = K(kb..kb+7), elems 8..15 = K(kb+16..kb+23), kb = (lane>=16)?8:0
// B layout: lane = N, elems 0..7 = K(kb..kb+7), elems 8..15 = K(kb+16..kb+23), kb = (lane>=16)?8:0 ; requires B stored [N][K]
// C layout: VGPR r -> M = r + ((lane>=16)?8:0), N = lane&15
// ---------------------------------------------------------------------------
__device__ __forceinline__ v8f vzero8() {
  v8f z;
#pragma unroll
  for (int e = 0; e < 8; ++e) z[e] = 0.f;
  return z;
}

__device__ __forceinline__ v16h frag_a(const _Float16* p, int ld, int row0, int k0) {
  const int lane = threadIdx.x & 31;
  const int m  = row0 + (lane & 15);
  const int kb = k0 + ((lane >> 4) << 3);
  const _Float16* r = p + (size_t)m * ld + kb;
  union { v16h v; v8h h[2]; } u;
  u.h[0] = *(const v8h*)(r);
  u.h[1] = *(const v8h*)(r + 16);
  return u.v;
}

__device__ __forceinline__ v16h frag_b(const _Float16* bt, int ld, int n0, int k0) {
  const int lane = threadIdx.x & 31;
  const int n  = n0 + (lane & 15);
  const int kb = k0 + ((lane >> 4) << 3);
  const _Float16* r = bt + (size_t)n * ld + kb;
  union { v16h v; v8h h[2]; } u;
  u.h[0] = *(const v8h*)(r);
  u.h[1] = *(const v8h*)(r + 16);
  return u.v;
}

__device__ __forceinline__ v8f wmma_f16(v16h a, v16h b, v8f c) {
  return __builtin_amdgcn_wmma_f32_16x16x32_f16(false, a, false, b, (short)0, c, false, false);
}

// Async copy of 16 bytes (global -> LDS), tracked by ASYNCcnt.
__device__ __forceinline__ void async_b128(_Float16* lds_dst, const _Float16* gsrc) {
  const unsigned dsoff = (unsigned)(size_t)lds_dst;   // shared aperture low 32 bits = LDS offset
  asm volatile("global_load_async_to_lds_b128 %0, %1, off"
               :: "v"(dsoff), "v"(gsrc) : "memory");
}

// ---------------------------------------------------------------------------
// Generic GEMM: C[M,N] = A[M,K](f16,row-major) * Bt[N,K](f16) + bias, opt ReLU
// block = 256 threads (8 waves, 2x4), block tile 128x256, wave tile 64x64
// 16 WMMAs per 16 b128 loads per lane per k-step (~85 FLOP/B from L2).
// ---------------------------------------------------------------------------
__global__ __launch_bounds__(256) void gemm_f16_kernel(
    const _Float16* __restrict__ A, const _Float16* __restrict__ Bt,
    const float* __restrict__ bias, float* __restrict__ Cf,
    _Float16* __restrict__ Ch, int M, int N, int K, int relu)
{
  const int lane = threadIdx.x & 31;
  const int wave = threadIdx.x >> 5;
  const int row0 = blockIdx.x * 128 + (wave >> 2) * 64;
  const int col0 = blockIdx.y * 256 + (wave & 3) * 64;
  (void)M;

  v8f acc[4][4];
#pragma unroll
  for (int i = 0; i < 4; ++i)
#pragma unroll
    for (int j = 0; j < 4; ++j) acc[i][j] = vzero8();

  for (int k0 = 0; k0 < K; k0 += 32) {
    v16h a[4];
#pragma unroll
    for (int mt = 0; mt < 4; ++mt) a[mt] = frag_a(A, K, row0 + mt * 16, k0);
#pragma unroll
    for (int nt = 0; nt < 4; ++nt) {
      const v16h b = frag_b(Bt, K, col0 + nt * 16, k0);
#pragma unroll
      for (int mt = 0; mt < 4; ++mt)
        acc[mt][nt] = wmma_f16(a[mt], b, acc[mt][nt]);
    }
  }

  const int nlo = lane & 15;
  const int rhi = (lane >> 4) * 8;
#pragma unroll
  for (int mt = 0; mt < 4; ++mt)
#pragma unroll
    for (int nt = 0; nt < 4; ++nt) {
      const int n = col0 + nt * 16 + nlo;
      const float bv = bias ? bias[n] : 0.f;
#pragma unroll
      for (int r = 0; r < 8; ++r) {
        const int m = row0 + mt * 16 + rhi + r;
        float v = acc[mt][nt][r] + bv;
        if (relu) v = fmaxf(v, 0.f);
        const size_t idx = (size_t)m * N + n;
        if (Cf) Cf[idx] = v;
        if (Ch) Ch[idx] = (_Float16)v;
      }
    }
}

// ---------------------------------------------------------------------------
// Flash attention: one wave = 16 query rows of one (b,h); streams keys 32 at a
// time with online softmax. K/V chunks staged cooperatively into LDS via
// async load-to-LDS (all 8 waves consume the same chunk). V is consumed
// pre-transposed [b][h][64][SEQ].
// ---------------------------------------------------------------------------
__global__ __launch_bounds__(256) void attn_kernel(
    const _Float16* __restrict__ Q, const _Float16* __restrict__ Km,
    const _Float16* __restrict__ Vt, _Float16* __restrict__ ctx_out)
{
  __shared__ __align__(16) _Float16 kbuf[32][64];     // [key_local][depth]
  __shared__ __align__(16) _Float16 vbuf[64][32];     // [depth][key_local]
  __shared__ __align__(16) _Float16 pbuf[8][16][32];  // per-wave P transpose
  const int tid  = threadIdx.x;
  const int lane = tid & 31;
  const int wave = tid >> 5;
  const int bh = blockIdx.y;
  const int b = bh >> 4, h = bh & 15;
  const int q0 = blockIdx.x * 128 + wave * 16;
  const _Float16* Qb = Q  + (size_t)b * SEQ * D_MODEL + h * DEPTH;
  const _Float16* Kb = Km + (size_t)b * SEQ * D_MODEL + h * DEPTH;
  const _Float16* Vb = Vt + (size_t)bh * DEPTH * SEQ;

  const v16h qa0 = frag_a(Qb, D_MODEL, q0, 0);
  const v16h qa1 = frag_a(Qb, D_MODEL, q0, 32);

  v8f ctx[4];
#pragma unroll
  for (int nt = 0; nt < 4; ++nt) ctx[nt] = vzero8();

  float mrow[8], lrow[8];
#pragma unroll
  for (int r = 0; r < 8; ++r) { mrow[r] = -3.0e30f; lrow[r] = 0.f; }

  const float SC  = 0.125f;               // 1/sqrt(DEPTH)
  const float L2E = 1.4426950408889634f;  // log2(e)
  const int rhi = (lane >> 4) * 8;
  const int nlo = lane & 15;

  for (int kk = 0; kk < SEQ; kk += 32) {
    // --- cooperative async stage: K chunk (32x64) + V chunk (64x32), 16B/thread each
    async_b128(&kbuf[tid >> 3][(tid & 7) * 8],
               Kb + (size_t)(kk + (tid >> 3)) * D_MODEL + (tid & 7) * 8);
    async_b128(&vbuf[tid >> 2][(tid & 3) * 8],
               Vb + (size_t)(tid >> 2) * SEQ + kk + (tid & 3) * 8);
    asm volatile("s_wait_asynccnt 0x0" ::: "memory");
    __syncthreads();

    // --- logits: 16 q rows x 32 keys, K-dim = depth 64
    v8f lg0 = vzero8(), lg1 = vzero8();
    lg0 = wmma_f16(qa0, frag_b(&kbuf[0][0], 64, 0,  0),  lg0);
    lg0 = wmma_f16(qa1, frag_b(&kbuf[0][0], 64, 0,  32), lg0);
    lg1 = wmma_f16(qa0, frag_b(&kbuf[0][0], 64, 16, 0),  lg1);
    lg1 = wmma_f16(qa1, frag_b(&kbuf[0][0], 64, 16, 32), lg1);

    // --- online softmax (row stats live in the half-wave that owns the rows)
    float alpha[8];
#pragma unroll
    for (int r = 0; r < 8; ++r) {
      const float x0 = lg0[r] * SC, x1 = lg1[r] * SC;
      float mx = fmaxf(x0, x1);
#pragma unroll
      for (int s = 8; s >= 1; s >>= 1) mx = fmaxf(mx, __shfl_xor(mx, s, 32));
      const float mn = fmaxf(mrow[r], mx);
      alpha[r] = exp2f((mrow[r] - mn) * L2E);
      mrow[r] = mn;
      const float p0 = exp2f((x0 - mn) * L2E);
      const float p1 = exp2f((x1 - mn) * L2E);
      pbuf[wave][rhi + r][nlo]      = (_Float16)p0;
      pbuf[wave][rhi + r][16 + nlo] = (_Float16)p1;
      float s2 = p0 + p1;
#pragma unroll
      for (int s = 8; s >= 1; s >>= 1) s2 += __shfl_xor(s2, s, 32);
      lrow[r] = lrow[r] * alpha[r] + s2;
    }
#pragma unroll
    for (int nt = 0; nt < 4; ++nt)
#pragma unroll
      for (int r = 0; r < 8; ++r) ctx[nt][r] *= alpha[r];

    asm volatile("s_wait_dscnt 0x0" ::: "memory");   // P tile visible in LDS (same wave)

    // --- ctx += P(16x32) * V(32x64)
    const v16h pa = frag_a(&pbuf[wave][0][0], 32, 0, 0);
#pragma unroll
    for (int nt = 0; nt < 4; ++nt)
      ctx[nt] = wmma_f16(pa, frag_b(&vbuf[0][0], 32, nt * 16, 0), ctx[nt]);

    __syncthreads();   // everyone done reading kbuf/vbuf before next stage
  }

#pragma unroll
  for (int nt = 0; nt < 4; ++nt) {
    const int col = h * DEPTH + nt * 16 + nlo;
#pragma unroll
    for (int r = 0; r < 8; ++r) {
      const size_t tok = (size_t)b * SEQ + q0 + rhi + r;
      ctx_out[tok * D_MODEL + col] = (_Float16)(ctx[nt][r] / lrow[r]);
    }
  }
}

// ---------------------------------------------------------------------------
// out = LayerNorm(X + Y); fp32 math; optional f32 / f16 outputs. 1 block / row.
// ---------------------------------------------------------------------------
__global__ __launch_bounds__(256) void add_ln_kernel(
    const float* __restrict__ X, const float* __restrict__ Y,
    const float* __restrict__ gamma, const float* __restrict__ beta,
    float* __restrict__ outF, _Float16* __restrict__ outH)
{
  __shared__ float red[8];
  const int tid = threadIdx.x;
  const size_t base = (size_t)blockIdx.x * D_MODEL;
  float v[4];
  float s = 0.f;
#pragma unroll
  for (int i = 0; i < 4; ++i) {
    const int c = tid + i * 256;
    v[i] = X[base + c] + Y[base + c];
    s += v[i];
  }
#pragma unroll
  for (int m = 16; m >= 1; m >>= 1) s += __shfl_xor(s, m, 32);
  if ((tid & 31) == 0) red[tid >> 5] = s;
  __syncthreads();
  if (tid < 32) {
    float t = (tid < 8) ? red[tid] : 0.f;
#pragma unroll
    for (int m = 4; m >= 1; m >>= 1) t += __shfl_xor(t, m, 32);
    if (tid == 0) red[0] = t;
  }
  __syncthreads();
  const float mu = red[0] * (1.f / D_MODEL);
  __syncthreads();

  float q = 0.f;
#pragma unroll
  for (int i = 0; i < 4; ++i) { const float d = v[i] - mu; q += d * d; }
#pragma unroll
  for (int m = 16; m >= 1; m >>= 1) q += __shfl_xor(q, m, 32);
  if ((tid & 31) == 0) red[tid >> 5] = q;
  __syncthreads();
  if (tid < 32) {
    float t = (tid < 8) ? red[tid] : 0.f;
#pragma unroll
    for (int m = 4; m >= 1; m >>= 1) t += __shfl_xor(t, m, 32);
    if (tid == 0) red[0] = t;
  }
  __syncthreads();
  const float rs = rsqrtf(red[0] * (1.f / D_MODEL) + 1e-6f);
#pragma unroll
  for (int i = 0; i < 4; ++i) {
    const int c = tid + i * 256;
    const float o = (v[i] - mu) * rs * gamma[c] + beta[c];
    if (outF) outF[base + c] = o;
    if (outH) outH[base + c] = (_Float16)o;
  }
}

// ---------------------------------------------------------------------------
// conversion / transpose utilities
// ---------------------------------------------------------------------------
__global__ __launch_bounds__(256) void cvt_f16_kernel(const float* __restrict__ in,
                                                      _Float16* __restrict__ out, int n) {
  const int i = blockIdx.x * 256 + threadIdx.x;
  if (i < n) out[i] = (_Float16)in[i];
}

// in[K][N] (f32) -> out[N][K] (f16)
__global__ __launch_bounds__(256) void transpose_cvt_kernel(const float* __restrict__ in,
                                                            _Float16* __restrict__ out,
                                                            int K, int N) {
  const size_t i = (size_t)blockIdx.x * 256 + threadIdx.x;
  if (i < (size_t)K * N) {
    const int k = (int)(i / N), n = (int)(i % N);
    out[(size_t)n * K + k] = (_Float16)in[i];
  }
}

// V[tok][D_MODEL] -> Vt[b][h][DEPTH][SEQ]
__global__ __launch_bounds__(256) void transpose_v_kernel(const _Float16* __restrict__ V,
                                                          _Float16* __restrict__ Vt) {
  const size_t i = (size_t)blockIdx.x * 256 + threadIdx.x;
  if (i < (size_t)NTOK * D_MODEL) {
    const int tok = (int)(i >> 10), c = (int)(i & 1023);
    const int b = tok >> 11, s = tok & 2047;
    const int h = c >> 6, d = c & 63;
    Vt[(((size_t)(b * NHEAD + h)) * DEPTH + d) * SEQ + s] = V[i];
  }
}

// ---------------------------------------------------------------------------
extern "C" void kernel_launch(void* const* d_in, const int* in_sizes, int n_in,
                              void* d_out, int out_size, void* d_ws, size_t ws_size,
                              hipStream_t stream) {
  (void)in_sizes; (void)n_in; (void)out_size; (void)ws_size;
  const float* x     = (const float*)d_in[0];
  const float* wq    = (const float*)d_in[1];
  const float* bq    = (const float*)d_in[2];
  const float* wk    = (const float*)d_in[3];
  const float* bk    = (const float*)d_in[4];
  const float* wv    = (const float*)d_in[5];
  const float* bv    = (const float*)d_in[6];
  const float* wo    = (const float*)d_in[7];
  const float* bo    = (const float*)d_in[8];
  const float* w1    = (const float*)d_in[9];
  const float* b1    = (const float*)d_in[10];
  const float* w2    = (const float*)d_in[11];
  const float* b2    = (const float*)d_in[12];
  const float* ln1_g = (const float*)d_in[13];
  const float* ln1_b = (const float*)d_in[14];
  const float* ln2_g = (const float*)d_in[15];
  const float* ln2_b = (const float*)d_in[16];

  char* ws = (char*)d_ws;
  size_t off = 0;
  auto take = [&](size_t bytes) -> void* {
    void* p = (void*)(ws + off);
    off += (bytes + 255) & ~(size_t)255;
    return p;
  };
  const size_t DD = (size_t)D_MODEL * D_MODEL;   // 1M
  const size_t TD = (size_t)NTOK * D_MODEL;      // 4M
  const size_t DF = (size_t)D_MODEL * DFF;       // 4M

  _Float16* WQT  = (_Float16*)take(DD * 2);
  _Float16* WKT  = (_Float16*)take(DD * 2);
  _Float16* WVT  = (_Float16*)take(DD * 2);
  _Float16* WOT  = (_Float16*)take(DD * 2);
  _Float16* W1T  = (_Float16*)take(DF * 2);
  _Float16* W2T  = (_Float16*)take(DF * 2);
  _Float16* XH   = (_Float16*)take(TD * 2);
  _Float16* QH   = (_Float16*)take(TD * 2);       // |-- this 32MB region is
  _Float16* KH   = (_Float16*)take(TD * 2);       // |   reused as the FFN
  _Float16* VH   = (_Float16*)take(TD * 2);       // |   hidden buffer after
  _Float16* VT   = (_Float16*)take(TD * 2);       // |   attention is done
  _Float16* CTXH = (_Float16*)take(TD * 2);
  float*    TMPF = (float*)take(TD * 4);          // attn_out, later ffn_out
  float*    OUT1F= (float*)take(TD * 4);
  _Float16* OUT1H= (_Float16*)take(TD * 2);
  _Float16* HIDH = QH;                            // alias: NTOK x DFF f16 (32MB)

  const dim3 blk(256);

  // --- precision conversion (weights L2-resident thereafter) ---
  cvt_f16_kernel<<<(int)((TD + 255) / 256), blk, 0, stream>>>(x, XH, (int)TD);
  transpose_cvt_kernel<<<(int)((DD + 255) / 256), blk, 0, stream>>>(wq, WQT, D_MODEL, D_MODEL);
  transpose_cvt_kernel<<<(int)((DD + 255) / 256), blk, 0, stream>>>(wk, WKT, D_MODEL, D_MODEL);
  transpose_cvt_kernel<<<(int)((DD + 255) / 256), blk, 0, stream>>>(wv, WVT, D_MODEL, D_MODEL);
  transpose_cvt_kernel<<<(int)((DD + 255) / 256), blk, 0, stream>>>(wo, WOT, D_MODEL, D_MODEL);
  transpose_cvt_kernel<<<(int)((DF + 255) / 256), blk, 0, stream>>>(w1, W1T, D_MODEL, DFF);
  transpose_cvt_kernel<<<(int)((DF + 255) / 256), blk, 0, stream>>>(w2, W2T, DFF, D_MODEL);

  // --- QKV projections (WMMA) ---
  const dim3 gDD(NTOK / 128, D_MODEL / 256);
  gemm_f16_kernel<<<gDD, blk, 0, stream>>>(XH, WQT, bq, nullptr, QH, NTOK, D_MODEL, D_MODEL, 0);
  gemm_f16_kernel<<<gDD, blk, 0, stream>>>(XH, WKT, bk, nullptr, KH, NTOK, D_MODEL, D_MODEL, 0);
  gemm_f16_kernel<<<gDD, blk, 0, stream>>>(XH, WVT, bv, nullptr, VH, NTOK, D_MODEL, D_MODEL, 0);
  transpose_v_kernel<<<(int)((TD + 255) / 256), blk, 0, stream>>>(VH, VT);

  // --- flash attention (WMMA logits + ctx, async LDS staging, online softmax) ---
  attn_kernel<<<dim3(SEQ / 128, BATCH * NHEAD), blk, 0, stream>>>(QH, KH, VT, CTXH);

  // --- output projection + residual/LN1 ---
  gemm_f16_kernel<<<gDD, blk, 0, stream>>>(CTXH, WOT, bo, TMPF, nullptr, NTOK, D_MODEL, D_MODEL, 0);
  add_ln_kernel<<<NTOK, blk, 0, stream>>>(x, TMPF, ln1_g, ln1_b, OUT1F, OUT1H);

  // --- FFN ---
  const dim3 gDF(NTOK / 128, DFF / 256);
  gemm_f16_kernel<<<gDF, blk, 0, stream>>>(OUT1H, W1T, b1, nullptr, HIDH, NTOK, DFF, D_MODEL, 1);
  gemm_f16_kernel<<<gDD, blk, 0, stream>>>(HIDH, W2T, b2, TMPF, nullptr, NTOK, D_MODEL, DFF, 0);

  // --- residual/LN2 -> output ---
  add_ln_kernel<<<NTOK, blk, 0, stream>>>(OUT1F, TMPF, ln2_g, ln2_b, (float*)d_out, nullptr);
}
